// HQQSVDLinear_19619410608652
// MI455X (gfx1250) — 8-bit matrix operand order repaired and checked by reading, compiled, not run on hardware
//
#include <hip/hip_runtime.h>
#include <hip/hip_bf16.h>
#include <stdint.h>

#define OUT_F    4096
#define N_GROUPS 32
#define GROUP    128
#define IN_F     4096
#define RANK     128
#define TOKENS   4096

typedef __attribute__((ext_vector_type(8))) int v8i;
typedef __attribute__((ext_vector_type(4))) int v4i;
typedef __attribute__((ext_vector_type(8))) signed char v8c;

#ifndef __has_builtin
#define __has_builtin(x) 0
#endif

#if defined(__HIP_DEVICE_COMPILE__) && \
    __has_builtin(__builtin_amdgcn_global_load_async_to_lds_b128) && \
    __has_builtin(__builtin_amdgcn_s_wait_asynccnt)
#define HAVE_ASYNC_LDS 1
#else
#define HAVE_ASYNC_LDS 0
#endif

#if HAVE_ASYNC_LDS
typedef __attribute__((address_space(1))) v4i gv4i;
typedef __attribute__((address_space(3))) v4i lv4i;
__device__ __forceinline__ gv4i* to_as1(const void* p) {
  return (gv4i*)(uintptr_t)p;
}
__device__ __forceinline__ lv4i* to_as3(void* p) {
  return (lv4i*)(uint32_t)(uintptr_t)p;
}
#endif

#define OPB 4
__global__ __launch_bounds__(256) void hqq_wdequant(
    const int* __restrict__ Wq, const float* __restrict__ scale,
    const float* __restrict__ zp, const float* __restrict__ svd_up,
    const float* __restrict__ svd_down, signed char* __restrict__ wq8,
    float* __restrict__ scale_w)
{
  __shared__ float s_up[OPB][RANK];
  __shared__ float s_sc[OPB][N_GROUPS];
  __shared__ float s_zp[OPB][N_GROUPS];
  __shared__ float s_red[256];

  const int tid = threadIdx.x;
  const int obase = blockIdx.x * OPB;

  for (int idx = tid; idx < OPB * RANK; idx += 256) {
    int oo = idx >> 7, r = idx & (RANK - 1);
    s_up[oo][r] = svd_up[(size_t)(obase + oo) * RANK + r];
  }
  if (tid < OPB * N_GROUPS) {
    int oo = tid >> 5, g = tid & 31;
    s_sc[oo][g] = scale[(obase + oo) * N_GROUPS + g];
    s_zp[oo][g] = zp[(obase + oo) * N_GROUPS + g];
  }
  __syncthreads();

  float w[OPB][16];
  #pragma unroll
  for (int oo = 0; oo < OPB; oo++)
    #pragma unroll
    for (int j = 0; j < 16; j++) w[oo][j] = 0.0f;

  for (int r = 0; r < RANK; r++) {
    const float u0 = s_up[0][r], u1 = s_up[1][r], u2 = s_up[2][r], u3 = s_up[3][r];
    const float* dr = svd_down + (size_t)r * IN_F + tid;
    #pragma unroll
    for (int j = 0; j < 16; j++) {
      float d = dr[256 * j];
      w[0][j] = fmaf(u0, d, w[0][j]);
      w[1][j] = fmaf(u1, d, w[1][j]);
      w[2][j] = fmaf(u2, d, w[2][j]);
      w[3][j] = fmaf(u3, d, w[3][j]);
    }
  }

  float amax[OPB] = {0.0f, 0.0f, 0.0f, 0.0f};
  #pragma unroll
  for (int oo = 0; oo < OPB; oo++) {
    const int* wrow = Wq + (size_t)(obase + oo) * IN_F + tid;
    #pragma unroll
    for (int j = 0; j < 16; j++) {
      int i = tid + 256 * j;
      int g = i >> 7;
      float q = (float)wrow[256 * j];
      float v = (q - s_zp[oo][g]) * s_sc[oo][g] + w[oo][j];
      w[oo][j] = v;
      amax[oo] = fmaxf(amax[oo], fabsf(v));
    }
  }

  float rowmax[OPB];
  #pragma unroll
  for (int oo = 0; oo < OPB; oo++) {
    __syncthreads();
    s_red[tid] = amax[oo];
    __syncthreads();
    for (int s = 128; s > 0; s >>= 1) {
      if (tid < s) s_red[tid] = fmaxf(s_red[tid], s_red[tid + s]);
      __syncthreads();
    }
    rowmax[oo] = s_red[0];
  }

  #pragma unroll
  for (int oo = 0; oo < OPB; oo++) {
    const float m = rowmax[oo];
    const float inv = (m > 0.0f) ? (127.0f / m) : 0.0f;
    if (tid == 0) scale_w[obase + oo] = m / 127.0f;
    signed char* orow = wq8 + (size_t)(obase + oo) * IN_F + tid;
    #pragma unroll
    for (int j = 0; j < 16; j++) {
      float q = rintf(w[oo][j] * inv);
      q = fminf(fmaxf(q, -128.0f), 127.0f);
      orow[256 * j] = (signed char)(int)q;
    }
  }
}

__global__ __launch_bounds__(256) void hqq_xquant(
    const float* __restrict__ x, signed char* __restrict__ xq8,
    float* __restrict__ scale_x)
{
  __shared__ float s_red[256];
  const int tid = threadIdx.x;
  const int row = blockIdx.x;
  const float* xr = x + (size_t)row * IN_F + tid * 16;

  float v[16];
  float amax = 0.0f;
  const float4* xr4 = (const float4*)xr;
  #pragma unroll
  for (int j4 = 0; j4 < 4; j4++) {
    float4 f = xr4[j4];
    v[j4 * 4 + 0] = f.x; v[j4 * 4 + 1] = f.y;
    v[j4 * 4 + 2] = f.z; v[j4 * 4 + 3] = f.w;
  }
  #pragma unroll
  for (int j = 0; j < 16; j++) amax = fmaxf(amax, fabsf(v[j]));

  s_red[tid] = amax;
  __syncthreads();
  for (int s = 128; s > 0; s >>= 1) {
    if (tid < s) s_red[tid] = fmaxf(s_red[tid], s_red[tid + s]);
    __syncthreads();
  }
  const float m = s_red[0];
  if (tid == 0) scale_x[row] = m / 127.0f;
  const float inv = (m > 0.0f) ? (127.0f / m) : 0.0f;

  v8c lo, hi;
  #pragma unroll
  for (int j = 0; j < 8; j++) {
    float q = fminf(fmaxf(rintf(v[j] * inv), -128.0f), 127.0f);
    lo[j] = (signed char)(int)q;
  }
  #pragma unroll
  for (int j = 0; j < 8; j++) {
    float q = fminf(fmaxf(rintf(v[8 + j] * inv), -128.0f), 127.0f);
    hi[j] = (signed char)(int)q;
  }

  const int group = tid >> 2;
  const int c = tid & 3;
  signed char* base = xq8 + (size_t)row * IN_F + group * 64;
  *(v8c*)(base + (c & 1) * 32 + (c >> 1) * 16)     = lo;
  *(v8c*)(base + (c & 1) * 32 + (c >> 1) * 16 + 8) = hi;
}

#define KSTEP 128
#define NTILE (IN_F / KSTEP)

__global__ __launch_bounds__(256) void hqq_gemm_i8(
    const signed char* __restrict__ xq8, const signed char* __restrict__ wq8,
    const float* __restrict__ scale_x, const float* __restrict__ scale_w,
    const float* __restrict__ bias, float* __restrict__ out)
{
  const int tid  = threadIdx.x;
  const int bm   = blockIdx.x * 64;
  const int bn   = blockIdx.y * 128;
  const int wid  = tid >> 5;
  const int lane = tid & 31;
  const int lh   = lane & 15;
  const int hi   = lane >> 4;
  const int wm   = wid >> 2;
  const int wn   = wid & 3;

  v8i acc[2][2];
  #pragma unroll
  for (int mt = 0; mt < 2; mt++)
    #pragma unroll
    for (int nt = 0; nt < 2; nt++)
      #pragma unroll
      for (int e = 0; e < 8; e++) acc[mt][nt][e] = 0;

  const int a_row = tid >> 2;
  const int a_col = (tid & 3) * 32;
  const signed char* a_g = xq8 + (size_t)(bm + a_row) * IN_F + a_col;
  const int b_row = tid >> 1;
  const int b_col = (tid & 1) * 64;
  const signed char* b_g = wq8 + (size_t)(bn + b_row) * IN_F + b_col;

#if HAVE_ASYNC_LDS
  __shared__ __align__(16) signed char Alds[2][64 * KSTEP];
  __shared__ __align__(16) signed char Blds[2][128 * KSTEP];
#else
  __shared__ __align__(16) signed char Alds[1][64 * KSTEP];
  __shared__ __align__(16) signed char Blds[1][128 * KSTEP];
#endif

  auto compute_tile = [&](const signed char* Ab, const signed char* Bb) {
    #pragma unroll
    for (int ks = 0; ks < KSTEP; ks += 64) {
      v8i afrag[2];
      #pragma unroll
      for (int mt = 0; mt < 2; mt++) {
        const int mrow = wm * 32 + mt * 16 + lh;
        afrag[mt] = *(const v8i*)(&Ab[mrow * KSTEP + ks + hi * 32]);
      }
      v8i bfrag[2];
      #pragma unroll
      for (int nt = 0; nt < 2; nt++) {
        const int nrow = wn * 32 + nt * 16 + lh;
        v4i blo = *(const v4i*)(&Bb[nrow * KSTEP + ks + hi * 16]);
        v4i bhi = *(const v4i*)(&Bb[nrow * KSTEP + ks + 32 + hi * 16]);
        v8i b;
        b[0] = blo[0]; b[1] = blo[1]; b[2] = blo[2]; b[3] = blo[3];
        b[4] = bhi[0]; b[5] = bhi[1]; b[6] = bhi[2]; b[7] = bhi[3];
        bfrag[nt] = b;
      }
      #pragma unroll
      for (int mt = 0; mt < 2; mt++)
        #pragma unroll
        for (int nt = 0; nt < 2; nt++)
          acc[mt][nt] = __builtin_amdgcn_wmma_i32_16x16x64_iu8(
              true, afrag[mt], true, bfrag[nt], acc[mt][nt], false, false);
    }
  };

#if HAVE_ASYNC_LDS
  #define ISSUE_TILE(it_, buf_)                                                \
    do {                                                                       \
      const int kk_ = (it_) * KSTEP;                                           \
      __builtin_amdgcn_global_load_async_to_lds_b128(                          \
          to_as1(a_g + kk_),      to_as3(&Alds[buf_][a_row * KSTEP + a_col]),      0, 0); \
      __builtin_amdgcn_global_load_async_to_lds_b128(                          \
          to_as1(a_g + kk_ + 16), to_as3(&Alds[buf_][a_row * KSTEP + a_col + 16]), 0, 0); \
      __builtin_amdgcn_global_load_async_to_lds_b128(                          \
          to_as1(b_g + kk_),      to_as3(&Blds[buf_][b_row * KSTEP + b_col]),      0, 0); \
      __builtin_amdgcn_global_load_async_to_lds_b128(                          \
          to_as1(b_g + kk_ + 16), to_as3(&Blds[buf_][b_row * KSTEP + b_col + 16]), 0, 0); \
      __builtin_amdgcn_global_load_async_to_lds_b128(                          \
          to_as1(b_g + kk_ + 32), to_as3(&Blds[buf_][b_row * KSTEP + b_col + 32]), 0, 0); \
      __builtin_amdgcn_global_load_async_to_lds_b128(                          \
          to_as1(b_g + kk_ + 48), to_as3(&Blds[buf_][b_row * KSTEP + b_col + 48]), 0, 0); \
    } while (0)

  ISSUE_TILE(0, 0);
  ISSUE_TILE(1, 1);

  #pragma unroll 2
  for (int it = 0; it < NTILE - 2; ++it) {
    __builtin_amdgcn_s_wait_asynccnt(6);
    __syncthreads();
    compute_tile(Alds[it & 1], Blds[it & 1]);
    __syncthreads();
    ISSUE_TILE(it + 2, it & 1);
  }
  __builtin_amdgcn_s_wait_asynccnt(6);
  __syncthreads();
  compute_tile(Alds[(NTILE - 2) & 1], Blds[(NTILE - 2) & 1]);
  __builtin_amdgcn_s_wait_asynccnt(0);
  __syncthreads();
  compute_tile(Alds[(NTILE - 1) & 1], Blds[(NTILE - 1) & 1]);
  #undef ISSUE_TILE
#else
  for (int k0 = 0; k0 < IN_F; k0 += KSTEP) {
    *(v4i*)(&Alds[0][a_row * KSTEP + a_col])      = *(const v4i*)(a_g + k0);
    *(v4i*)(&Alds[0][a_row * KSTEP + a_col + 16]) = *(const v4i*)(a_g + k0 + 16);
    *(v4i*)(&Blds[0][b_row * KSTEP + b_col])      = *(const v4i*)(b_g + k0);
    *(v4i*)(&Blds[0][b_row * KSTEP + b_col + 16]) = *(const v4i*)(b_g + k0 + 16);
    *(v4i*)(&Blds[0][b_row * KSTEP + b_col + 32]) = *(const v4i*)(b_g + k0 + 32);
    *(v4i*)(&Blds[0][b_row * KSTEP + b_col + 48]) = *(const v4i*)(b_g + k0 + 48);
    __syncthreads();
    compute_tile(Alds[0], Blds[0]);
    __syncthreads();
  }
#endif

  #pragma unroll
  for (int mt = 0; mt < 2; mt++) {
    const int row_base = bm + wm * 32 + mt * 16 + hi * 8;
    #pragma unroll
    for (int nt = 0; nt < 2; nt++) {
      const int col = bn + wn * 32 + nt * 16 + lh;
      const float sw = scale_w[col];
      const float bb = bias[col];
      #pragma unroll
      for (int r = 0; r < 8; r++) {
        const int row = row_base + r;
        out[(size_t)row * OUT_F + col] =
            (float)acc[mt][nt][r] * scale_x[row] * sw + bb;
      }
    }
  }
}

extern "C" void kernel_launch(void* const* d_in, const int* in_sizes, int n_in,
                              void* d_out, int out_size, void* d_ws, size_t ws_size,
                              hipStream_t stream) {
  const float* x        = (const float*)d_in[0];
  const int*   Wq       = (const int*)  d_in[1];
  const float* scale    = (const float*)d_in[2];
  const float* zp       = (const float*)d_in[3];
  const float* svd_up   = (const float*)d_in[4];
  const float* svd_down = (const float*)d_in[5];
  const float* bias     = (const float*)d_in[6];
  float* out = (float*)d_out;

  char* ws = (char*)d_ws;
  signed char* wq8 = (signed char*)ws;
  signed char* xq8 = (signed char*)(ws + ((size_t)16 << 20));
  float* scale_w   = (float*)(ws + ((size_t)32 << 20));
  float* scale_x   = (float*)(ws + ((size_t)32 << 20) + (1 << 16));

  hqq_wdequant<<<OUT_F / OPB, 256, 0, stream>>>(Wq, scale, zp, svd_up, svd_down,
                                                wq8, scale_w);
  hqq_xquant<<<TOKENS, 256, 0, stream>>>(x, xq8, scale_x);
  hqq_gemm_i8<<<dim3(TOKENS / 64, OUT_F / 128), 256, 0, stream>>>(
      xq8, wq8, scale_x, scale_w, bias, out);
}
